// LinearAttention_expand_APE_75041668596125
// MI455X (gfx1250) — compile-verified, matrix operand order checked by reading, not run on hardware
//
#include <hip/hip_runtime.h>
#include <hip/hip_bf16.h>

typedef __attribute__((ext_vector_type(16))) _Float16 v16h;
typedef __attribute__((ext_vector_type(8)))  float    v8f;
typedef __attribute__((ext_vector_type(4)))  float    f32x4;

#define B_      8
#define DIM_    256
#define HEADS_  8
#define DHEAD_  64
#define RED_    16
#define H_      64
#define W_      64
#define INNER_  512
#define NPIX    4096   /* H*W */
#define NPOOL   256    /* RED*RED */
#define SCALE_  0.125f /* DHEAD^-0.5 */

static __device__ __forceinline__ int lane_id() { return (int)(threadIdx.x & 31u); }

// ---------------------------------------------------------------------------
// CDNA5 wave32 WMMA fragment helpers (v_wmma_f32_16x16x32_f16).
// A (16x32 f16): lane l -> row m=l&15; lanes<16 hold K {0..7,16..23},
//                lanes>=16 hold +8. a[i]: k = kb+i (i<8), kb+8+i (i>=8).
// B (32x16 f16): lane l -> col n=l&15; same K split as A.
// C/D (16x16 f32): lane l -> col n=l&15; c[j] -> row mb+j, mb=8*(l>=16).
// ---------------------------------------------------------------------------

// A element addr = p + m*ld + k (row-major A). Contiguous K -> 4x b128 loads.
// Requires p rows 16B-aligned and k0 a multiple of 8 (true everywhere here).
__device__ __forceinline__ v16h ld_a_row(const float* p, size_t ld, int m0, int k0) {
  int l = lane_id();
  const float* r = p + (size_t)(m0 + (l & 15)) * ld + (size_t)(k0 + ((l & 16) >> 1));
  const f32x4* r4 = (const f32x4*)r;
  f32x4 u0 = r4[0], u1 = r4[1], u2 = r4[4], u3 = r4[5];  // k: 0..7, 16..23
#pragma clang diagnostic push
#pragma clang diagnostic ignored "-Wuninitialized"
  v16h a;
#pragma unroll
  for (int i = 0; i < 4; ++i) {
    a[i]      = (_Float16)u0[i];
    a[4 + i]  = (_Float16)u1[i];
    a[8 + i]  = (_Float16)u2[i];
    a[12 + i] = (_Float16)u3[i];
  }
#pragma clang diagnostic pop
  return a;
}

// A element addr = p + k*ldk + m  (k-major A, e.g. channel-strided Q)
__device__ __forceinline__ v16h ld_a_col(const float* p, size_t ldk, int m0, int k0) {
  int l = lane_id();
  const float* r = p + (size_t)(m0 + (l & 15)) + (size_t)(k0 + ((l & 16) >> 1)) * ldk;
#pragma clang diagnostic push
#pragma clang diagnostic ignored "-Wuninitialized"
  v16h a;
#pragma unroll
  for (int i = 0; i < 8; ++i)  a[i] = (_Float16)r[(size_t)i * ldk];
#pragma unroll
  for (int i = 8; i < 16; ++i) a[i] = (_Float16)r[(size_t)(8 + i) * ldk];
#pragma clang diagnostic pop
  return a;
}

// B element addr = p + k*ldk + n  (row-major B, rows indexed by K; lane-coalesced)
__device__ __forceinline__ v16h ld_b_row(const float* p, size_t ldk, int k0, int n0) {
  int l = lane_id();
  const float* r = p + (size_t)(k0 + ((l & 16) >> 1)) * ldk + (size_t)(n0 + (l & 15));
#pragma clang diagnostic push
#pragma clang diagnostic ignored "-Wuninitialized"
  v16h b;
#pragma unroll
  for (int i = 0; i < 8; ++i)  b[i] = (_Float16)r[(size_t)i * ldk];
#pragma unroll
  for (int i = 8; i < 16; ++i) b[i] = (_Float16)r[(size_t)(8 + i) * ldk];
#pragma clang diagnostic pop
  return b;
}

// B element addr = p + n*ldn + k (n-major B, e.g. V). Contiguous K -> 4x b128.
__device__ __forceinline__ v16h ld_b_col(const float* p, size_t ldn, int k0, int n0) {
  int l = lane_id();
  const float* r = p + (size_t)(n0 + (l & 15)) * ldn + (size_t)(k0 + ((l & 16) >> 1));
  const f32x4* r4 = (const f32x4*)r;
  f32x4 u0 = r4[0], u1 = r4[1], u2 = r4[4], u3 = r4[5];
#pragma clang diagnostic push
#pragma clang diagnostic ignored "-Wuninitialized"
  v16h b;
#pragma unroll
  for (int i = 0; i < 4; ++i) {
    b[i]      = (_Float16)u0[i];
    b[4 + i]  = (_Float16)u1[i];
    b[8 + i]  = (_Float16)u2[i];
    b[12 + i] = (_Float16)u3[i];
  }
#pragma clang diagnostic pop
  return b;
}

__device__ __forceinline__ v8f wmma_f16(v16h a, v16h b, v8f c) {
  return __builtin_amdgcn_wmma_f32_16x16x32_f16(false, a, false, b, (short)0, c,
                                                false, false);
}

// ---------------------------------------------------------------------------
// Kernel 1/5: depthwise 3x3, pad 1, (B, C, 64, 64) -> same.
// ---------------------------------------------------------------------------
__global__ void __launch_bounds__(256)
dw3x3(const float* __restrict__ in, const float* __restrict__ w,
      float* __restrict__ out, int C) {
  int idx = blockIdx.x * 256 + threadIdx.x;
  int x = idx & 63, y = (idx >> 6) & 63;
  int c = (idx >> 12) % C;
  int b = idx / (C * NPIX);
  const float* wp = w + (size_t)c * 9;
  const float* ip = in + ((size_t)b * C + c) * NPIX;
  float acc = 0.f;
#pragma unroll
  for (int ky = 0; ky < 3; ++ky) {
    int yy = y + ky - 1;
    if (yy < 0 || yy > 63) continue;
#pragma unroll
    for (int kx = 0; kx < 3; ++kx) {
      int xx = x + kx - 1;
      if (xx < 0 || xx > 63) continue;
      acc += ip[yy * 64 + xx] * wp[ky * 3 + kx];
    }
  }
  out[idx] = acc;
}

// ---------------------------------------------------------------------------
// Kernels 2/6: pointwise 1x1 conv as WMMA GEMM.
// out[b, m, n] = sum_k wgt[m, k] * act[b, k, n],   n over NPIX pixels.
// Block = 8 waves, each wave owns one 16x16 output tile (block tile 16x128).
// ---------------------------------------------------------------------------
__global__ void __launch_bounds__(256)
pw_gemm(const float* __restrict__ wgt, const float* __restrict__ act,
        float* __restrict__ out, int M, int K) {
  int wave = (int)(threadIdx.x >> 5);
  int l = lane_id();
  int ntb = NPIX / 128;                 // 32 n-block tiles
  int bid = blockIdx.x;
  int ntile = bid % ntb;
  int mtile = (bid / ntb) % (M / 16);
  int b = bid / (ntb * (M / 16));
  int m0 = mtile * 16;
  int n0 = ntile * 128 + wave * 16;
  const float* actb = act + (size_t)b * (size_t)K * NPIX;

  v8f c = {};
  for (int k0 = 0; k0 < K; k0 += 32) {
    if (k0 + 32 < K)  // pull next activation k-tile toward L2 (global_prefetch_b8)
      __builtin_prefetch(actb + (size_t)(k0 + 32) * NPIX + n0 + l, 0, 1);
    v16h a = ld_a_row(wgt, (size_t)K, m0, k0);      // weights (M x K) row-major
    v16h bm = ld_b_row(actb, (size_t)NPIX, k0, n0); // activations (K x NPIX)
    c = wmma_f16(a, bm, c);
  }
  // C store: lane -> col n=l&15, rows mb..mb+7
  float* dst = out + (size_t)b * (size_t)M * NPIX;
  int n = n0 + (l & 15);
  int mb = m0 + ((l & 16) >> 1);
#pragma unroll
  for (int j = 0; j < 8; ++j) dst[(size_t)(mb + j) * NPIX + n] = c[j];
}

// ---------------------------------------------------------------------------
// Kernel 3: adaptive max pool 64 -> 16 (4x4, stride 4) on K and V slices of qkv.
// qkv layout (B, 3*INNER, 64, 64); K = ch 512..1023, V = ch 1024..1535.
// ---------------------------------------------------------------------------
__global__ void __launch_bounds__(256)
pool4(const float* __restrict__ qkv, float* __restrict__ kp, float* __restrict__ vp) {
  int idx = blockIdx.x * 256 + threadIdx.x;
  int sel = idx / (B_ * INNER_ * NPOOL);     // 0 -> K, 1 -> V
  int r = idx % (B_ * INNER_ * NPOOL);
  int px = r & 15, py = (r >> 4) & 15;
  int c = (r >> 8) & (INNER_ - 1);
  int b = r >> 17;                           // / (512*256)
  const float* src = qkv + ((size_t)b * (3 * INNER_) + (sel ? 1024 : 512) + c) * NPIX;
  float m = -3.0e38f;
#pragma unroll
  for (int dy = 0; dy < 4; ++dy) {
    f32x4 v = *(const f32x4*)(src + (py * 4 + dy) * 64 + px * 4);  // 16B aligned
#pragma unroll
    for (int dx = 0; dx < 4; ++dx) m = fmaxf(m, v[dx]);
  }
  (sel ? vp : kp)[r] = m;
}

// ---------------------------------------------------------------------------
// Kernel 4: attention. One block per (b, head, 64-row i-tile). 8 waves.
// Stage 0: async-copy pooled K^T slice (64x256 f32) into LDS
//          (global_load_async_to_lds_b128, ASYNCcnt).
// Phase 1: S = Q Kt (64x256) via WMMA -> LDS.
// Phase 2: softmax((S + ape) * scale) in LDS; NT-stream attn to d_out region 2.
// Phase 3: O = P V (64x64) via WMMA -> ov in (B, INNER, H, W) layout.
// Head channel mapping from to_heads: ch = d*HEADS + head.
// ---------------------------------------------------------------------------
__global__ void __launch_bounds__(256)
attn_kernel(const float* __restrict__ qkv, const float* __restrict__ kp,
            const float* __restrict__ vp, const float* __restrict__ ape,
            float* __restrict__ attn_out, float* __restrict__ ov) {
  __shared__ float kt_sh[DHEAD_ * NPOOL];     // 64 KB: K^T slice, [d][j]
  __shared__ float sc[64 * 256];              // 64 KB: score tile

  int bid = blockIdx.x;
  int itile = bid & 63;
  int head = (bid >> 6) & 7;
  int b = bid >> 9;
  int ibase = itile * 64;
  int wave = (int)(threadIdx.x >> 5);
  int l = lane_id();

  const float* qbase = qkv + ((size_t)b * (3 * INNER_) + head) * NPIX;  // + d*8*NPIX + i
  const float* kbase = kp + ((size_t)b * INNER_ + head) * NPOOL;        // + d*8*NPOOL + j
  const float* vbase = vp + ((size_t)b * INNER_ + head) * NPOOL;

  // ---- Stage 0: async copy K^T (b,head) slice into LDS, 16B granules ----
  for (int t = (int)threadIdx.x; t < (DHEAD_ * NPOOL) / 4; t += 256) {
    int e = t * 4;                             // element index; j runs contiguous
    const float* gp = kbase + (size_t)(e >> 8) * ((size_t)HEADS_ * NPOOL) + (e & 255);
    unsigned long long ga = (unsigned long long)(uintptr_t)gp;
    unsigned int la = (unsigned int)(uintptr_t)&kt_sh[e];  // LDS byte offset
    asm volatile("global_load_async_to_lds_b128 %0, %1, off"
                 :: "v"(la), "v"(ga) : "memory");
  }
  asm volatile("s_wait_asynccnt 0x0" ::: "memory");
  __syncthreads();

  // ---- Phase 1: scores. 4 m-tiles x 16 n-tiles, wave w -> (w&3, 8 n-tiles) ----
  {
    int m0 = (wave & 3) * 16;
    int nh = wave >> 2;
    // Q fragment: A[m][k=d], element = qbase + d*(8*NPIX) + (ibase+m); K=64 -> 2 steps
    v16h a0 = ld_a_col(qbase, (size_t)HEADS_ * NPIX, ibase + m0, 0);
    v16h a1 = ld_a_col(qbase, (size_t)HEADS_ * NPIX, ibase + m0, 32);
#pragma unroll
    for (int jt = 0; jt < 8; ++jt) {
      int n0 = (nh * 8 + jt) * 16;
      // Kt fragment from LDS: B[k=d][n=j], element = kt_sh[d*256 + j]
      v16h b0 = ld_b_row(kt_sh, (size_t)NPOOL, 0, n0);
      v16h b1 = ld_b_row(kt_sh, (size_t)NPOOL, 32, n0);
      v8f c = {};
      c = wmma_f16(a0, b0, c);
      c = wmma_f16(a1, b1, c);
      int n = n0 + (l & 15);
      int mb = m0 + ((l & 16) >> 1);
#pragma unroll
      for (int j = 0; j < 8; ++j) sc[(mb + j) * 256 + n] = c[j];
    }
  }
  __syncthreads();

  // ---- Phase 2: softmax over j (256) with APE bias ----
  if (threadIdx.x < 64) {
    int i = ibase + (int)threadIdx.x;
    float* row = sc + (size_t)threadIdx.x * 256;
    const float* ar = ape + (size_t)i * NPOOL;
    float mx = -3.0e38f;
    for (int j = 0; j < 256; ++j) {
      float v = (row[j] + ar[j]) * SCALE_;
      row[j] = v;
      mx = fmaxf(mx, v);
    }
    float s = 0.f;
    for (int j = 0; j < 256; ++j) {
      float e = __expf(row[j] - mx);
      row[j] = e;
      s += e;
    }
    float inv = 1.f / s;
    for (int j = 0; j < 256; ++j) row[j] *= inv;
  }
  __syncthreads();

  // Non-temporal stream of attn tile (write-once, 256 MB total: keep out of L2).
  {
    float* abase = attn_out + (((size_t)b * HEADS_ + head) * NPIX + ibase) * NPOOL;
    for (int t = (int)threadIdx.x; t < 64 * 256; t += 256)
      __builtin_nontemporal_store(sc[t], abase + t);
  }

  // ---- Phase 3: O = P V. M=64, N=64(d), K=256(j). 16 tiles / 8 waves ----
  {
    float* ovbase = ov + ((size_t)b * INNER_ + head) * NPIX;  // + d*8*NPIX + i
    int m0 = (wave & 3) * 16;
#pragma unroll
    for (int rep = 0; rep < 2; ++rep) {
      int n0 = ((wave >> 2) + rep * 2) * 16;
      v8f c = {};
      for (int k0 = 0; k0 < 256; k0 += 32) {
        v16h ap = ld_a_row(sc, (size_t)256, m0, k0);               // P from LDS
        v16h bv = ld_b_col(vbase, (size_t)HEADS_ * NPOOL, k0, n0); // V: p + d*2048 + j
        c = wmma_f16(ap, bv, c);
      }
      int n = n0 + (l & 15);
      int mb = m0 + ((l & 16) >> 1);
#pragma unroll
      for (int j = 0; j < 8; ++j)
        ovbase[(size_t)n * ((size_t)HEADS_ * NPIX) + (size_t)(ibase + mb + j)] = c[j];
    }
  }
}

// ---------------------------------------------------------------------------
// Launcher: x, w_qkv_dw, w_qkv_pw, w_out_dw, w_out_pw, ape  ->  (out, attn)
// ---------------------------------------------------------------------------
extern "C" void kernel_launch(void* const* d_in, const int* in_sizes, int n_in,
                              void* d_out, int out_size, void* d_ws, size_t ws_size,
                              hipStream_t stream) {
  (void)in_sizes; (void)n_in; (void)out_size; (void)ws_size;
  const float* x        = (const float*)d_in[0];
  const float* w_qkv_dw = (const float*)d_in[1];
  const float* w_qkv_pw = (const float*)d_in[2];
  const float* w_out_dw = (const float*)d_in[3];
  const float* w_out_pw = (const float*)d_in[4];
  const float* ape      = (const float*)d_in[5];

  float* out      = (float*)d_out;                        // (B, DIM, 64, 64)
  float* attn_out = out + (size_t)B_ * DIM_ * NPIX;       // (B, HEADS, 4096, 256)

  float* ws = (float*)d_ws;
  size_t o = 0;
  float* y1  = ws + o; o += (size_t)B_ * DIM_ * NPIX;       // dw(x)
  float* qkv = ws + o; o += (size_t)B_ * 3 * INNER_ * NPIX; // pw qkv
  float* kpb = ws + o; o += (size_t)B_ * INNER_ * NPOOL;    // pooled K
  float* vpb = ws + o; o += (size_t)B_ * INNER_ * NPOOL;    // pooled V
  float* ov  = ws + o; o += (size_t)B_ * INNER_ * NPIX;     // attention out
  float* y2  = ws + o;                                      // dw(ov)

  // 1) depthwise on x (256 ch)
  dw3x3<<<(B_ * DIM_ * NPIX) / 256, 256, 0, stream>>>(x, w_qkv_dw, y1, DIM_);
  // 2) pointwise 256 -> 1536 (WMMA GEMM)
  pw_gemm<<<B_ * ((3 * INNER_) / 16) * (NPIX / 128), 256, 0, stream>>>(
      w_qkv_pw, y1, qkv, 3 * INNER_, DIM_);
  // 3) 4x4 max pool on K,V
  pool4<<<(2 * B_ * INNER_ * NPOOL) / 256, 256, 0, stream>>>(qkv, kpb, vpb);
  // 4) attention (async K staging + WMMA + LDS softmax), writes attn region
  attn_kernel<<<B_ * HEADS_ * (NPIX / 64), 256, 0, stream>>>(
      qkv, kpb, vpb, ape, attn_out, ov);
  // 5) depthwise on attention output (512 ch)
  dw3x3<<<(B_ * INNER_ * NPIX) / 256, 256, 0, stream>>>(ov, w_out_dw, y2, INNER_);
  // 6) pointwise 512 -> 256 (WMMA GEMM) into out region of d_out
  pw_gemm<<<B_ * (DIM_ / 16) * (NPIX / 128), 256, 0, stream>>>(
      w_out_pw, y2, out, DIM_, INNER_);
}
